// Block_12635793785563
// MI455X (gfx1250) — matrix operand order repaired and checked by reading, not compiled, not run on hardware
//
#include <hip/hip_runtime.h>
#include <math.h>

typedef __attribute__((ext_vector_type(16))) __bf16 bf16x16;
typedef __attribute__((ext_vector_type(8)))  float  f32x8;
typedef __attribute__((ext_vector_type(4)))  unsigned int u32x4;
typedef __attribute__((ext_vector_type(8)))  int          i32x8;
typedef __attribute__((ext_vector_type(4)))  int          i32x4;

#define BATCH   8
#define SEQ     1024
#define CH      768
#define NH      12
#define HD      64
#define HID     3072
#define MROWS   (BATCH * SEQ)   // 8192

// ---------------------------------------------------------------------------
// TDM: async DMA of a [32 x 128] bf16 tile of Bt[K][N] into LDS.
// D# per CDNA5 ISA (2-D tensor, groups 2/3 disabled).
// ---------------------------------------------------------------------------
__device__ __forceinline__
void tdm_load_b_tile(const __bf16* gsrc, unsigned int ldsOff, int N, int K)
{
    unsigned long long ga = (unsigned long long)(const void*)gsrc;
    u32x4 g0 = { 1u,                                    // count=1, no gather
                 ldsOff,                                // lds_addr (bytes)
                 (unsigned int)ga,                      // global_addr[31:0]
                 (unsigned int)((ga >> 32) & 0x01FFFFFFu) | 0x80000000u }; // [56:32] | type=2
    i32x8 g1 = { 0x00010000,                            // data_size=1 (2B), mask=0
                 (int)((N & 0xFFFF) << 16),             // tensor_dim0[15:0]
                 (int)(((N >> 16) & 0xFFFF) | ((K & 0xFFFF) << 16)),   // dim0 hi | dim1 lo
                 (int)(((K >> 16) & 0xFFFF) | (128 << 16)),            // dim1 hi | tile_dim0=128
                 32,                                    // tile_dim1=32 (tile_dim2=0)
                 N,                                     // tensor_dim0_stride[31:0]
                 0, 0 };
    i32x4 z4 = { 0, 0, 0, 0 };
#if defined(__clang_major__) && (__clang_major__ >= 23)
    i32x8 z8 = { 0, 0, 0, 0, 0, 0, 0, 0 };
    __builtin_amdgcn_tensor_load_to_lds(g0, g1, z4, z4, z8, 0);
#else
    __builtin_amdgcn_tensor_load_to_lds(g0, g1, z4, z4, 0);
#endif
}

// ---------------------------------------------------------------------------
// Weight convert + transpose:  W[N][K] f32  ->  Wt[K][N] bf16  (write-coalesced)
// ---------------------------------------------------------------------------
__global__ __launch_bounds__(256)
void transpose_bf16_kernel(const float* __restrict__ W, __bf16* __restrict__ Wt,
                           int Nn, int Kk)
{
    size_t idx = (size_t)blockIdx.x * 256 + threadIdx.x;
    size_t total = (size_t)Nn * Kk;
    if (idx >= total) return;
    int k = (int)(idx / Nn);
    int n = (int)(idx % Nn);
    Wt[idx] = (__bf16)W[(size_t)n * Kk + k];
}

// ---------------------------------------------------------------------------
// LayerNorm(row of 768) -> bf16. One block of 256 threads per row (3 elems/thr).
// ---------------------------------------------------------------------------
__global__ __launch_bounds__(256)
void layernorm_bf16_kernel(const float* __restrict__ x, const float* __restrict__ w,
                           const float* __restrict__ bb, __bf16* __restrict__ out)
{
    const int row = blockIdx.x;
    const int tid = threadIdx.x;
    const float* xr = x + (size_t)row * CH;
    float v0 = xr[tid], v1 = xr[tid + 256], v2 = xr[tid + 512];
    float s  = v0 + v1 + v2;
    float sq = v0 * v0 + v1 * v1 + v2 * v2;
    #pragma unroll
    for (int m = 1; m < 32; m <<= 1) { s += __shfl_xor(s, m); sq += __shfl_xor(sq, m); }
    __shared__ float sh_s[8], sh_q[8];
    int wave = tid >> 5, lane = tid & 31;
    if (lane == 0) { sh_s[wave] = s; sh_q[wave] = sq; }
    __syncthreads();
    float S = 0.f, Q = 0.f;
    #pragma unroll
    for (int i = 0; i < 8; i++) { S += sh_s[i]; Q += sh_q[i]; }
    float mean = S * (1.f / CH);
    float var  = Q * (1.f / CH) - mean * mean;
    float rstd = rsqrtf(var + 1e-5f);
    __bf16* orow = out + (size_t)row * CH;
    orow[tid]       = (__bf16)((v0 - mean) * rstd * w[tid]       + bb[tid]);
    orow[tid + 256] = (__bf16)((v1 - mean) * rstd * w[tid + 256] + bb[tid + 256]);
    orow[tid + 512] = (__bf16)((v2 - mean) * rstd * w[tid + 512] + bb[tid + 512]);
}

// ---------------------------------------------------------------------------
// WMMA GEMM:  Out[M][N] = A[M][K](bf16) * Bt[K][N](bf16) + bias, fused epilogue.
// Block tile 128x128, 8 waves (4 along M, 2 along N), wave tile 32x64.
// A: direct global fragment loads (row=rowBase+(lane&15), 32B contiguous).
// B: TDM double-buffered [32x128] LDS tiles; per-lane ds fragment:
//    column n = lane&15, rows k = 16*(lane>>4)+s (stride 256B between elements).
// ---------------------------------------------------------------------------
enum { MODE_QKV = 0, MODE_PROJ = 1, MODE_GELU = 2, MODE_FC2 = 3 };

template <int MODE>
__global__ __launch_bounds__(256)
void gemm_wmma_kernel(const __bf16* __restrict__ A, const __bf16* __restrict__ Bt,
                      const float* __restrict__ bias, const float* __restrict__ resid,
                      float* __restrict__ outF, __bf16* __restrict__ outB,
                      __bf16* __restrict__ qOut, __bf16* __restrict__ ktOut,
                      __bf16* __restrict__ vOut, int M, int N, int K)
{
    __shared__ __bf16 ldsB[2][32 * 128];   // double-buffered B tile, 2 x 8 KB
    const int lane  = threadIdx.x & 31;
    const int wave  = threadIdx.x >> 5;
    const int wm    = wave & 3, wn = wave >> 2;
    const int rowBase = blockIdx.x * 128 + wm * 32;
    const int colBase = blockIdx.y * 128 + wn * 64;
    const int lhalf = lane >> 4, lmod = lane & 15;

    f32x8 acc[2][4];
    #pragma unroll
    for (int mi = 0; mi < 2; mi++)
        #pragma unroll
        for (int ni = 0; ni < 4; ni++) acc[mi][ni] = (f32x8){};

    const __bf16* aPtr0 = A + (size_t)(rowBase + lmod) * K + lhalf * 16;
    const __bf16* aPtr1 = aPtr0 + (size_t)16 * K;
    const __bf16* bTile = Bt + (size_t)blockIdx.y * 128;   // + k0*N per step
    const unsigned int ldsBase0 = (unsigned int)(size_t)&ldsB[0][0];
    const unsigned int ldsBase1 = (unsigned int)(size_t)&ldsB[1][0];

    // Prologue: DMA first B tile (wave 0 only; EXEC-independent TDM op)
    if (wave == 0) {
        tdm_load_b_tile(bTile, ldsBase0, N, K);
        __builtin_amdgcn_s_wait_tensorcnt(0);
    }
    __syncthreads();

    #pragma unroll 2
    for (int k0 = 0; k0 < K; k0 += 32) {
        const int cur = (k0 >> 5) & 1;
        // Kick off next tile DMA while this one is consumed
        if (wave == 0 && (k0 + 32) < K)
            tdm_load_b_tile(bTile + (size_t)(k0 + 32) * N,
                            cur ? ldsBase0 : ldsBase1, N, K);
        if ((k0 + 128) < K) {   // uniform: no EXEC divergence
            __builtin_prefetch(aPtr0 + k0 + 128, 0, 1);
            __builtin_prefetch(aPtr1 + k0 + 128, 0, 1);
        }
        bf16x16 a[2], bfr[4];
        a[0] = *(const bf16x16*)(aPtr0 + k0);
        a[1] = *(const bf16x16*)(aPtr1 + k0);
        const __bf16* bk = &ldsB[cur][lhalf * 16 * 128 + wn * 64 + lmod];
        #pragma unroll
        for (int ni = 0; ni < 4; ni++)
            #pragma unroll
            for (int s = 0; s < 16; s++) bfr[ni][s] = bk[s * 128 + ni * 16];
        #pragma unroll
        for (int mi = 0; mi < 2; mi++)
            #pragma unroll
            for (int ni = 0; ni < 4; ni++)
                acc[mi][ni] = __builtin_amdgcn_wmma_f32_16x16x32_bf16(
                    false, a[mi], false, bfr[ni], (short)0, acc[mi][ni], false, false);
        // Publish next buffer / protect current one before next iteration
        if (wave == 0) __builtin_amdgcn_s_wait_tensorcnt(0);
        __syncthreads();
    }

    // Epilogue. D layout: element i -> (m = i + 8*(lane>>4), n = lane&15)
    #pragma unroll
    for (int mi = 0; mi < 2; mi++) {
        #pragma unroll
        for (int ni = 0; ni < 4; ni++) {
            #pragma unroll
            for (int i = 0; i < 8; i++) {
                int row = rowBase + mi * 16 + lhalf * 8 + i;
                int col = colBase + ni * 16 + lmod;
                float v = acc[mi][ni][i] + bias[col];
                if constexpr (MODE == MODE_PROJ || MODE == MODE_FC2) {
                    outF[(size_t)row * N + col] = v + resid[(size_t)row * N + col];
                } else if constexpr (MODE == MODE_GELU) {
                    float g = 0.5f * v * (1.f + erff(v * 0.70710678118654752f));
                    outB[(size_t)row * N + col] = (__bf16)g;
                } else { // MODE_QKV: scatter to Q[b,h,t,d], Kt[b,h,d,t], V[b,h,t,d]
                    int b = row >> 10, t = row & 1023;
                    if (col < CH) {
                        int h = col >> 6, dd = col & 63;
                        qOut[(((size_t)(b * NH + h)) * SEQ + t) * HD + dd] = (__bf16)v;
                    } else if (col < 2 * CH) {
                        int c2 = col - CH; int h = c2 >> 6, dd = c2 & 63;
                        ktOut[(((size_t)(b * NH + h)) * HD + dd) * SEQ + t] = (__bf16)v;
                    } else {
                        int c2 = col - 2 * CH; int h = c2 >> 6, dd = c2 & 63;
                        vOut[(((size_t)(b * NH + h)) * SEQ + t) * HD + dd] = (__bf16)v;
                    }
                }
            }
        }
    }
}

// ---------------------------------------------------------------------------
// Flash attention. Grid (SEQ/128, B*NH); 8 waves/block, 16 q-rows per wave.
// 64-key blocks: S via 8 WMMAs, one online-softmax reduction per row per block,
// O += P*V via 8 WMMAs. P relaid D-layout -> A-layout through per-wave LDS.
// ---------------------------------------------------------------------------
__global__ __launch_bounds__(256)
void attn_kernel(const __bf16* __restrict__ Q, const __bf16* __restrict__ Kt,
                 const __bf16* __restrict__ V, __bf16* __restrict__ O)
{
    __shared__ __bf16 ldsP[8 * 16 * 64];       // 8 waves * 16x64 bf16 = 16 KB
    const int lane  = threadIdx.x & 31;
    const int wave  = threadIdx.x >> 5;
    const int bh    = blockIdx.y;              // b*NH + h
    const int qRow0 = blockIdx.x * 128 + wave * 16;
    const int lhalf = lane >> 4, lmod = lane & 15;
    const int b = bh / NH, h = bh % NH;

    // Q A-fragments for the two 32-wide d-steps (held in VGPRs whole kernel)
    const __bf16* qBase = Q + ((size_t)bh * SEQ + qRow0 + lmod) * HD + lhalf * 16;
    bf16x16 aq0 = *(const bf16x16*)(qBase);
    bf16x16 aq1 = *(const bf16x16*)(qBase + 32);

    float mstate[8], lstate[8];
    f32x8 accO[4];
    #pragma unroll
    for (int i = 0; i < 8; i++) { mstate[i] = -INFINITY; lstate[i] = 0.f; }
    #pragma unroll
    for (int ni = 0; ni < 4; ni++) accO[ni] = (f32x8){};

    __bf16* pPatch = &ldsP[wave * 1024];

    for (int kb = 0; kb < SEQ; kb += 64) {
        // ---- S tiles: 16 q x 64 keys (4 subtiles of 16 keys) ----
        f32x8 s[4];
        #pragma unroll
        for (int nt = 0; nt < 4; nt++) s[nt] = (f32x8){};
        const __bf16* ktp = Kt + ((size_t)bh * HD + lhalf * 16) * SEQ + kb + lmod;
        #pragma unroll
        for (int nt = 0; nt < 4; nt++) {
            bf16x16 b0, b1;                                            // d 0..31 / d 32..63
            #pragma unroll
            for (int s = 0; s < 16; s++) {
                b0[s] = ktp[(size_t)s * SEQ + nt * 16];
                b1[s] = ktp[(size_t)(32 + s) * SEQ + nt * 16];
            }
            s[nt] = __builtin_amdgcn_wmma_f32_16x16x32_bf16(false, aq0, false, b0, (short)0, s[nt], false, false);
            s[nt] = __builtin_amdgcn_wmma_f32_16x16x32_bf16(false, aq1, false, b1, (short)0, s[nt], false, false);
        }

        // ---- online softmax (one cross-lane tree per row per 64 keys) ----
        float p[4][8], alpha[8];
        #pragma unroll
        for (int i = 0; i < 8; i++) {
            float sv0 = s[0][i] * 0.125f;    // 1/sqrt(64)
            float sv1 = s[1][i] * 0.125f;
            float sv2 = s[2][i] * 0.125f;
            float sv3 = s[3][i] * 0.125f;
            float v = fmaxf(fmaxf(sv0, sv1), fmaxf(sv2, sv3));
            v = fmaxf(v, __shfl_xor(v, 1));
            v = fmaxf(v, __shfl_xor(v, 2));
            v = fmaxf(v, __shfl_xor(v, 4));
            v = fmaxf(v, __shfl_xor(v, 8));
            float mn = fmaxf(mstate[i], v);
            float a  = __expf(mstate[i] - mn);
            float e0 = __expf(sv0 - mn), e1 = __expf(sv1 - mn);
            float e2 = __expf(sv2 - mn), e3 = __expf(sv3 - mn);
            float rs = (e0 + e1) + (e2 + e3);
            rs += __shfl_xor(rs, 1); rs += __shfl_xor(rs, 2);
            rs += __shfl_xor(rs, 4); rs += __shfl_xor(rs, 8);
            lstate[i] = lstate[i] * a + rs;
            mstate[i] = mn;
            alpha[i]  = a;
            p[0][i] = e0; p[1][i] = e1; p[2][i] = e2; p[3][i] = e3;
        }
        #pragma unroll
        for (int ni = 0; ni < 4; ni++)
            #pragma unroll
            for (int i = 0; i < 8; i++) accO[ni][i] *= alpha[i];

        // ---- relayout P through per-wave LDS patch (in-order DS, no barrier) ----
        #pragma unroll
        for (int i = 0; i < 8; i++) {
            int r = i + 8 * lhalf;
            #pragma unroll
            for (int nt = 0; nt < 4; nt++)
                pPatch[r * 64 + nt * 16 + lmod] = (__bf16)p[nt][i];
        }
        bf16x16 pf0 = *(const bf16x16*)&pPatch[lmod * 64 + lhalf * 16];      // keys 0..31
        bf16x16 pf1 = *(const bf16x16*)&pPatch[lmod * 64 + 32 + lhalf * 16]; // keys 32..63

        // ---- O += P * V (two 32-key reduction steps) ----
        #pragma unroll
        for (int ks = 0; ks < 2; ks++) {
            const __bf16* vp = V + ((size_t)bh * SEQ + kb + ks * 32 + lhalf * 16) * HD + lmod;
            bf16x16 pf = ks ? pf1 : pf0;
            #pragma unroll
            for (int ni = 0; ni < 4; ni++) {
                bf16x16 bv;
                #pragma unroll
                for (int s = 0; s < 16; s++) bv[s] = vp[s * HD + ni * 16];
                accO[ni] = __builtin_amdgcn_wmma_f32_16x16x32_bf16(
                    false, pf, false, bv, (short)0, accO[ni], false, false);
            }
        }
    }

    // ---- normalize + store O[b, token, h*64+d] (bf16) ----
    #pragma unroll
    for (int i = 0; i < 8; i++) {
        float inv = 1.f / lstate[i];
        int token = qRow0 + 8 * lhalf + i;
        __bf16* orow = O + ((size_t)b * SEQ + token) * CH + h * HD;
        #pragma unroll
        for (int ni = 0; ni < 4; ni++)
            orow[ni * 16 + lmod] = (__bf16)(accO[ni][i] * inv);
    }
}

// ---------------------------------------------------------------------------
extern "C" void kernel_launch(void* const* d_in, const int* in_sizes, int n_in,
                              void* d_out, int out_size, void* d_ws, size_t ws_size,
                              hipStream_t stream)
{
    const float* x      = (const float*)d_in[0];
    const float* ln1_w  = (const float*)d_in[1];
    const float* ln1_b  = (const float*)d_in[2];
    const float* qkv_w  = (const float*)d_in[3];
    const float* qkv_b  = (const float*)d_in[4];
    const float* proj_w = (const float*)d_in[5];
    const float* proj_b = (const float*)d_in[6];
    const float* ln2_w  = (const float*)d_in[7];
    const float* ln2_b  = (const float*)d_in[8];
    const float* fc1_w  = (const float*)d_in[9];
    const float* fc1_b  = (const float*)d_in[10];
    const float* fc2_w  = (const float*)d_in[11];
    const float* fc2_b  = (const float*)d_in[12];

    char* p = (char*)d_ws;
    auto alloc = [&](size_t bytes) -> void* {
        void* r = (void*)p; p += (bytes + 255) & ~(size_t)255; return r;
    };
    __bf16* qkvWt = (__bf16*)alloc((size_t)CH * 3 * CH * 2);    // [768][2304]
    __bf16* projWt= (__bf16*)alloc((size_t)CH * CH * 2);        // [768][768]
    __bf16* fc1Wt = (__bf16*)alloc((size_t)CH * HID * 2);       // [768][3072]
    __bf16* fc2Wt = (__bf16*)alloc((size_t)HID * CH * 2);       // [3072][768]
    __bf16* h1    = (__bf16*)alloc((size_t)MROWS * CH * 2);
    __bf16* qB    = (__bf16*)alloc((size_t)MROWS * CH * 2);
    __bf16* ktB   = (__bf16*)alloc((size_t)MROWS * CH * 2);
    __bf16* vB    = (__bf16*)alloc((size_t)MROWS * CH * 2);
    __bf16* oB    = (__bf16*)alloc((size_t)MROWS * CH * 2);
    float*  x1    = (float*) alloc((size_t)MROWS * CH * 4);
    __bf16* h2    = (__bf16*)alloc((size_t)MROWS * CH * 2);
    __bf16* h3    = (__bf16*)alloc((size_t)MROWS * HID * 2);

    auto tgrid = [](size_t total) { return (unsigned)((total + 255) / 256); };

    // 1) weight convert + transpose to bf16 [K][N]
    transpose_bf16_kernel<<<tgrid((size_t)3 * CH * CH), 256, 0, stream>>>(qkv_w,  qkvWt, 3 * CH, CH);
    transpose_bf16_kernel<<<tgrid((size_t)CH * CH),     256, 0, stream>>>(proj_w, projWt, CH,    CH);
    transpose_bf16_kernel<<<tgrid((size_t)HID * CH),    256, 0, stream>>>(fc1_w,  fc1Wt,  HID,   CH);
    transpose_bf16_kernel<<<tgrid((size_t)CH * HID),    256, 0, stream>>>(fc2_w,  fc2Wt,  CH,    HID);

    // 2) LN1
    layernorm_bf16_kernel<<<MROWS, 256, 0, stream>>>(x, ln1_w, ln1_b, h1);

    // 3) QKV GEMM (8192 x 2304 x 768), scatter q / k^T / v
    gemm_wmma_kernel<MODE_QKV><<<dim3(MROWS / 128, (3 * CH) / 128), 256, 0, stream>>>(
        h1, qkvWt, qkv_b, nullptr, nullptr, nullptr, qB, ktB, vB, MROWS, 3 * CH, CH);

    // 4) attention
    attn_kernel<<<dim3(SEQ / 128, BATCH * NH), 256, 0, stream>>>(qB, ktB, vB, oB);

    // 5) proj GEMM + residual -> x1 (f32)
    gemm_wmma_kernel<MODE_PROJ><<<dim3(MROWS / 128, CH / 128), 256, 0, stream>>>(
        oB, projWt, proj_b, x, x1, nullptr, nullptr, nullptr, nullptr, MROWS, CH, CH);

    // 6) LN2
    layernorm_bf16_kernel<<<MROWS, 256, 0, stream>>>(x1, ln2_w, ln2_b, h2);

    // 7) fc1 GEMM + exact GELU -> h3 (bf16)
    gemm_wmma_kernel<MODE_GELU><<<dim3(MROWS / 128, HID / 128), 256, 0, stream>>>(
        h2, fc1Wt, fc1_b, nullptr, nullptr, h3, nullptr, nullptr, nullptr, MROWS, HID, CH);

    // 8) fc2 GEMM + residual -> d_out (f32)
    gemm_wmma_kernel<MODE_FC2><<<dim3(MROWS / 128, CH / 128), 256, 0, stream>>>(
        h3, fc2Wt, fc2_b, x1, (float*)d_out, nullptr, nullptr, nullptr, nullptr, MROWS, CH, HID);
}
